// GCNLayer_6219112645324
// MI455X (gfx1250) — matrix operand order repaired and checked by reading, not compiled, not run on hardware
//
#include <hip/hip_runtime.h>
#include <hip/hip_bf16.h>

// ---------------------------------------------------------------------------
// Relational GAT layer, B=8, N=1024, F=128, 12 attention relations + self.
// Strategy: bf16 WMMA (16x16x32) for all GEMMs, f32 softmax math,
// adjacency streamed once from HBM (software-pipelined into registers),
// h matrices L2-resident in bf16 (24 MB << 192 MB L2).
// ---------------------------------------------------------------------------

#define BATCH 8
#define NN    1024
#define FF    128
#define NREL  12

typedef __attribute__((ext_vector_type(16))) __bf16 bf16x16;
typedef __attribute__((ext_vector_type(8)))  __bf16 bf16x8;
typedef __attribute__((ext_vector_type(8)))  float  f32x8;

// ---- WMMA fragment helpers (contiguous k order, same in A and B, wave32) ----------

// A operand: 16x32 bf16 tile, row-major in LDS with row stride 32.
// lane m=lane&15, half=lane>>4:
//   elements 0..7  <-> K = 16*half + 0..7       (VGPR0-3)
//   elements 8..15 <-> K = 16*half + 8 + 0..7   (VGPR4-7)
__device__ __forceinline__ bf16x16 load_a_lds(const __bf16* Wl, int sr, int lane) {
    int m = lane & 15, half = lane >> 4;
    const __bf16* row = Wl + (sr * 16 + m) * 32;
    bf16x8 lo = *(const bf16x8*)(row + 16 * half);
    bf16x8 hi = *(const bf16x8*)(row + 16 * half + 8);
    bf16x16 a;
#pragma unroll
    for (int i = 0; i < 8; ++i) { a[i] = lo[i]; a[8 + i] = hi[i]; }
    return a;
}

// B operand: 32x16 bf16 tile from transposed-h storage Vt[F][N] bf16.
// lane n=lane&15, khalf=lane>>4: element e <-> K = 16*khalf + e  (contiguous!)
__device__ __forceinline__ bf16x16 load_b_g(const __bf16* Vt, int feat0, int c0, int lane) {
    int n = lane & 15, khalf = lane >> 4;
    return *(const bf16x16*)(Vt + (size_t)(feat0 + n) * NN + c0 + 16 * khalf);
}

// ---------------------------------------------------------------------------
// K1: the 13 input GEMMs  out[64x128] = x[64x128] @ W.T + b
// grid = (rowblk=16, matrix=13, batch=8), 256 threads (8 waves).
// m<12 -> store bf16 TRANSPOSED into Vt[b][rel][feat][row]
// m==12 (self path) -> store f32 row-major into out_init.
// ---------------------------------------------------------------------------
__global__ __launch_bounds__(256) void k1_gemm(
    const float* __restrict__ x,
    const float* __restrict__ fc_w,     const float* __restrict__ fc_b,
    const float* __restrict__ fwd_fc_w, const float* __restrict__ fwd_fc_b,
    const float* __restrict__ bwd_fc_w, const float* __restrict__ bwd_fc_b,
    const float* __restrict__ self_w,   const float* __restrict__ self_b,
    const float* __restrict__ bias_v,
    __bf16* __restrict__ Vt, float* __restrict__ out_init)
{
    const int t = threadIdx.x, lane = t & 31, wave = t >> 5;
    const int sr = wave & 3, sf = wave >> 2;        // row-slab 16, feat-slab 64
    const int R0 = blockIdx.x * 64;
    const int m  = blockIdx.y;                      // which weight matrix
    const int bat = blockIdx.z;

    const float* W; const float* bb; int rel = -1;
    if (m < 6)       { W = fc_w + m * FF * FF;            bb = fc_b + m * FF;            rel = m; }
    else if (m < 9)  { int j = m - 6; W = fwd_fc_w + j * FF * FF; bb = fwd_fc_b + j * FF; rel = 6 + 2 * j; }
    else if (m < 12) { int j = m - 9; W = bwd_fc_w + j * FF * FF; bb = bwd_fc_b + j * FF; rel = 7 + 2 * j; }
    else             { W = self_w;                         bb = self_b; }

    const float* xb = x + (size_t)bat * NN * FF;

    f32x8 acc[4] = {};
#pragma unroll
    for (int kt = 0; kt < 4; ++kt) {                // K = 128 in chunks of 32
        // A fragment from x (f32 -> bf16 on the fly)
        int mm = lane & 15, half = lane >> 4;
        const float* xr = xb + (size_t)(R0 + sr * 16 + mm) * FF + kt * 32;
        bf16x16 afrag;
#pragma unroll
        for (int i = 0; i < 8; ++i) {
            afrag[i]     = (__bf16)xr[16 * half + i];
            afrag[8 + i] = (__bf16)xr[16 * half + 8 + i];
        }
        int n = lane & 15, khalf = lane >> 4;
#pragma unroll
        for (int f = 0; f < 4; ++f) {
            // B(k, feat) = W[feat][k]  -> 16 contiguous f32 of the weight row
            const float* wr = W + (size_t)(sf * 64 + f * 16 + n) * FF + kt * 32 + 16 * khalf;
            bf16x16 bfrag;
#pragma unroll
            for (int e = 0; e < 16; ++e) bfrag[e] = (__bf16)wr[e];
            acc[f] = __builtin_amdgcn_wmma_f32_16x16x32_bf16(
                false, afrag, false, bfrag, (short)0, acc[f], false, false);
        }
    }

    // store: C element v <-> row = v + 8*(lane>=16), col = lane&15
    int n = lane & 15, half = lane >> 4;
#pragma unroll
    for (int f = 0; f < 4; ++f) {
        int feat = sf * 64 + f * 16 + n;
        if (m == 12) {
#pragma unroll
            for (int v = 0; v < 8; ++v) {
                int row = R0 + sr * 16 + v + 8 * half;
                out_init[((size_t)bat * NN + row) * FF + feat] = acc[f][v] + bb[feat] + bias_v[feat];
            }
        } else {
            // rows v are consecutive -> one 16B vector store into transposed Vt
            bf16x8 pk;
#pragma unroll
            for (int v = 0; v < 8; ++v) pk[v] = (__bf16)(acc[f][v] + bb[feat]);
            __bf16* dst = Vt + ((size_t)(bat * NREL + rel) * FF + feat) * NN
                            + R0 + sr * 16 + 8 * half;
            *(bf16x8*)dst = pk;
        }
    }
}

// ---------------------------------------------------------------------------
// K2: per-relation attention score vectors hl/hr (length N each).
// Relations >=6 use the stale h5 (faithful to reference). Bias folded into hl.
// grid = B*NREL*4 blocks of 256 (one row per thread).
// ---------------------------------------------------------------------------
__global__ __launch_bounds__(256) void k2_scores(
    const __bf16* __restrict__ Vt,
    const float* __restrict__ att_w,     const float* __restrict__ att_b,
    const float* __restrict__ fwd_att_w, const float* __restrict__ fwd_att_b,
    const float* __restrict__ bwd_att_w, const float* __restrict__ bwd_att_b,
    float* __restrict__ hl_g, float* __restrict__ hr_g)
{
    int idx  = blockIdx.x;
    int part = idx & 3;
    int rel  = (idx >> 2) % NREL;
    int bat  = idx / (4 * NREL);
    int i    = part * 256 + threadIdx.x;

    int src = (rel < 6) ? rel : 5;                  // stale h5 for fwd/bwd
    const __bf16* H = Vt + (size_t)(bat * NREL + src) * FF * NN;

    const float* wv; const float* bv;
    if (rel < 6)        { wv = att_w + rel * 2 * FF;               bv = att_b + rel; }
    else if (!(rel & 1)) { int j = (rel - 6) >> 1; wv = fwd_att_w + j * 2 * FF; bv = fwd_att_b + j; }
    else                 { int j = (rel - 7) >> 1; wv = bwd_att_w + j * 2 * FF; bv = bwd_att_b + j; }

    float sl = 0.f, sr = 0.f;
#pragma unroll 4
    for (int f = 0; f < FF; ++f) {
        float hv = (float)H[(size_t)f * NN + i];    // coalesced across lanes
        sl += hv * wv[f];
        sr += hv * wv[FF + f];
    }
    hl_g[(bat * NREL + rel) * NN + i] = sl + *bv;   // bias folded into hl
    hr_g[(bat * NREL + rel) * NN + i] = sr;
}

// ---------------------------------------------------------------------------
// K0: extract diagonals of adjs[3..5] (relations 3-5 use outer(d,d)).
// ---------------------------------------------------------------------------
__global__ void k0_diag(const float* __restrict__ adjs, float* __restrict__ diag)
{
    int idx = blockIdx.x * blockDim.x + threadIdx.x;
    if (idx >= BATCH * 3 * NN) return;
    int i  = idx & (NN - 1);
    int jj = (idx >> 10) % 3;
    int bat = idx / (3 * NN);
    diag[idx] = adjs[((size_t)(bat * 9 + 3 + jj) * NN + i) * NN + i];
}

// ---------------------------------------------------------------------------
// K3: attention-weighted aggregation.  Workgroup = (64-row block, batch).
// For each of 12 relations, stream 32-wide column tiles:
//   w_ij = adj_ij>0 ? exp(lrelu(hl_i + hr_j)) : 1       (dense softmax!)
//   num += W @ V  (bf16 WMMA),  den += rowsum(W)
// Adjacency loads for tile ct+1 are issued while tile ct's WMMAs run
// (register software pipeline -> HBM latency hidden even at 1 block/WGP).
// out_lds += num * scale / den ; final: relu(out_lds + self_term).
// grid = (16, 8), 256 threads (8 waves, wave=(sr,sf) 16x64 output slab).
// ---------------------------------------------------------------------------
__global__ __launch_bounds__(256) void k3_attn(
    const float*  __restrict__ adjs,
    const __bf16* __restrict__ Vt,
    const float*  __restrict__ out_init,
    const float*  __restrict__ hl_g, const float* __restrict__ hr_g,
    const float*  __restrict__ diag,
    float* __restrict__ out)
{
    const int t = threadIdx.x, lane = t & 31, wave = t >> 5;
    const int sr = wave & 3, sf = wave >> 2;
    const int R0  = blockIdx.x * 64;
    const int bat = blockIdx.y;

    __shared__ __bf16 Wl[64 * 32];        // 4 KB  current W tile (bf16)
    __shared__ float  out_lds[64 * FF];   // 32 KB accumulated output
    __shared__ float  den_lds[64];

    for (int i = t; i < 64 * FF; i += 256) out_lds[i] = 0.f;

    for (int rel = 0; rel < NREL; ++rel) {
        // kind: 0 plain adj, 1 outer(diag,diag), 2 fwd(adj), 3 bwd(adj^T)
        int kind; const float* adj = nullptr; const float* dj = nullptr; float scale;
        if (rel < 3)      { kind = 0; adj = adjs + (size_t)(bat * 9 + rel) * NN * NN; scale = 1.f; }
        else if (rel < 6) { kind = 1; dj = diag + (bat * 3 + (rel - 3)) * NN;        scale = 1.f; }
        else {
            int j = (rel - 6) >> 1;
            kind  = (rel & 1) ? 3 : 2;
            adj   = adjs + (size_t)(bat * 9 + 6 + j) * NN * NN;
            scale = 0.5f;
        }
        const float*  hl = hl_g + (bat * NREL + rel) * NN;
        const float*  hr = hr_g + (bat * NREL + rel) * NN;
        const __bf16* V  = Vt + (size_t)(bat * NREL + rel) * FF * NN;

        f32x8 acc[4] = {};
        float den = 0.f;

        // per-relation invariants: this thread's row scores + tile-0 prefetch
        float pf[8]  = {};
        float hlr[8] = {};
        float hlv = 0.f;
        if (kind != 3) {
            int rbase = t >> 5;
#pragma unroll
            for (int i = 0; i < 8; ++i) hlr[i] = hl[R0 + rbase + 8 * i];
            if (kind != 1) {
                int col = t & 31;
#pragma unroll
                for (int i = 0; i < 8; ++i)
                    pf[i] = adj[(size_t)(R0 + rbase + 8 * i) * NN + col];
            }
        } else {
            int m = t & 63, kk = t >> 6;
            hlv = hl[R0 + m];
#pragma unroll
            for (int i = 0; i < 8; ++i)
                pf[i] = adj[(size_t)(kk + 4 * i) * NN + R0 + m];
        }

        for (int ct = 0; ct < 32; ++ct) {
            const int c0 = ct * 32;
            __syncthreads();                 // Wl of previous tile fully consumed
            // ---- build 64x32 W tile from prefetched adjacency ----
            if (kind != 3) {
                int col = t & 31, rbase = t >> 5;
                float hrv  = hr[c0 + col];
                float dcol = (kind == 1) ? dj[c0 + col] : 0.f;
#pragma unroll
                for (int i = 0; i < 8; ++i) {
                    int row = rbase + 8 * i;
                    float a = (kind == 1) ? dj[R0 + row] * dcol : pf[i];
                    float w = 1.0f;                          // exp(lrelu(0))
                    if (a > 0.f) {
                        float e = hlr[i] + hrv;
                        w = __expf(e > 0.f ? e : 0.01f * e);
                    }
                    Wl[row * 32 + col] = (__bf16)w;
                }
                // issue next tile's HBM loads; they fly during den+WMMA below
                if (kind != 1 && ct < 31) {
#pragma unroll
                    for (int i = 0; i < 8; ++i)
                        pf[i] = adj[(size_t)(R0 + rbase + 8 * i) * NN + c0 + 32 + col];
                }
            } else {
                // transposed adjacency: adj rows were read coalesced along m
                int m = t & 63, kk = t >> 6;
#pragma unroll
                for (int i = 0; i < 8; ++i) {
                    int k = kk + 4 * i;
                    float a = pf[i];
                    float w = 1.0f;
                    if (a > 0.f) {
                        float e = hlv + hr[c0 + k];
                        w = __expf(e > 0.f ? e : 0.01f * e);
                    }
                    Wl[m * 32 + k] = (__bf16)w;
                }
                if (ct < 31) {
#pragma unroll
                    for (int i = 0; i < 8; ++i)
                        pf[i] = adj[(size_t)(c0 + 32 + kk + 4 * i) * NN + R0 + m];
                }
            }
            __syncthreads();
            // ---- softmax denominator (same bf16 values as WMMA sees) ----
            if (t < 64) {
                const __bf16* wrow = Wl + t * 32;
                float s = 0.f;
#pragma unroll
                for (int k = 0; k < 32; ++k) s += (float)wrow[k];
                den += s;
            }
            // ---- num += W @ V via bf16 WMMA ----
            bf16x16 afrag = load_a_lds(Wl, sr, lane);
#pragma unroll
            for (int f = 0; f < 4; ++f) {
                bf16x16 bfrag = load_b_g(V, sf * 64 + f * 16, c0, lane);
                acc[f] = __builtin_amdgcn_wmma_f32_16x16x32_bf16(
                    false, afrag, false, bfrag, (short)0, acc[f], false, false);
            }
        }
        if (t < 64) den_lds[t] = scale / den;   // one reciprocal per row
        __syncthreads();
        // ---- out += num * (scale/den) ----
        {
            int n = lane & 15, half = lane >> 4;
#pragma unroll
            for (int f = 0; f < 4; ++f) {
                int feat = sf * 64 + f * 16 + n;
#pragma unroll
                for (int v = 0; v < 8; ++v) {
                    int row = sr * 16 + v + 8 * half;
                    out_lds[row * FF + feat] += acc[f][v] * den_lds[row];
                }
            }
        }
    }
    __syncthreads();
    // ---- fuse self term + ReLU, store ----
    for (int i = t; i < 64 * FF; i += 256) {
        int row = i >> 7, feat = i & (FF - 1);
        size_t gi = ((size_t)bat * NN + R0 + row) * FF + feat;
        float v = out_lds[i] + out_init[gi];
        out[gi] = v > 0.f ? v : 0.f;
    }
}

// ---------------------------------------------------------------------------
extern "C" void kernel_launch(void* const* d_in, const int* in_sizes, int n_in,
                              void* d_out, int out_size, void* d_ws, size_t ws_size,
                              hipStream_t stream) {
    const float* x         = (const float*)d_in[0];
    const float* adjs      = (const float*)d_in[1];
    const float* fc_w      = (const float*)d_in[2];
    const float* fc_b      = (const float*)d_in[3];
    const float* att_w     = (const float*)d_in[4];
    const float* att_b     = (const float*)d_in[5];
    const float* fwd_fc_w  = (const float*)d_in[6];
    const float* fwd_fc_b  = (const float*)d_in[7];
    const float* fwd_att_w = (const float*)d_in[8];
    const float* fwd_att_b = (const float*)d_in[9];
    const float* bwd_fc_w  = (const float*)d_in[10];
    const float* bwd_fc_b  = (const float*)d_in[11];
    const float* bwd_att_w = (const float*)d_in[12];
    const float* bwd_att_b = (const float*)d_in[13];
    const float* self_w    = (const float*)d_in[14];
    const float* self_b    = (const float*)d_in[15];
    const float* bias_v    = (const float*)d_in[16];
    float* out = (float*)d_out;

    // workspace layout (~29 MB total)
    char* ws = (char*)d_ws;
    size_t off = 0;
    __bf16* Vt      = (__bf16*)(ws + off); off += (size_t)BATCH * NREL * FF * NN * sizeof(__bf16);
    float*  out_ini = (float*) (ws + off); off += (size_t)BATCH * NN * FF * sizeof(float);
    float*  hl_g    = (float*) (ws + off); off += (size_t)BATCH * NREL * NN * sizeof(float);
    float*  hr_g    = (float*) (ws + off); off += (size_t)BATCH * NREL * NN * sizeof(float);
    float*  diag    = (float*) (ws + off); off += (size_t)BATCH * 3 * NN * sizeof(float);
    (void)off; (void)ws_size; (void)in_sizes; (void)n_in; (void)out_size;

    // 1) all 13 input GEMMs (bf16 WMMA), transposed bf16 h + f32 self term
    k1_gemm<<<dim3(NN / 64, 13, BATCH), 256, 0, stream>>>(
        x, fc_w, fc_b, fwd_fc_w, fwd_fc_b, bwd_fc_w, bwd_fc_b,
        self_w, self_b, bias_v, Vt, out_ini);

    // 2) attention score vectors
    k2_scores<<<dim3(BATCH * NREL * 4), 256, 0, stream>>>(
        Vt, att_w, att_b, fwd_att_w, fwd_att_b, bwd_att_w, bwd_att_b, hl_g, hr_g);

    // 3) diagonals for relations 3-5
    k0_diag<<<dim3((BATCH * 3 * NN + 255) / 256), 256, 0, stream>>>(adjs, diag);

    // 4) dense-softmax attention aggregation (bf16 WMMA), fused ReLU
    k3_attn<<<dim3(NN / 64, BATCH), 256, 0, stream>>>(
        adjs, Vt, out_ini, hl_g, hr_g, diag, out);
}
